// YvSelectiveSSM_72112500900673
// MI455X (gfx1250) — matrix operand order repaired and checked by reading, not compiled, not run on hardware
//
#include <hip/hip_runtime.h>
#include <hip/hip_bf16.h>

typedef __attribute__((ext_vector_type(16))) _Float16 v16h;
typedef __attribute__((ext_vector_type(8)))  _Float16 v8h;
typedef __attribute__((ext_vector_type(8)))  float    v8f;

#define LDS_K      64
#define LDS_STRIDE 72     // 64 halves + 8 pad halves (16B) to stagger banks
#define LDS_MAT    (128 * LDS_STRIDE)
#define SCAN_G     16     // chunks over L
#define SCAN_LC    64     // steps per chunk (G * LC == L)

// ---------------------------------------------------------------- elementwise
__global__ __launch_bounds__(256) void cast_f16_kernel(const float* __restrict__ in,
                                                       _Float16* __restrict__ out, int n) {
  int i = blockIdx.x * 256 + threadIdx.x;
  if (i < n) out[i] = (_Float16)in[i];
}

// out[n*K + k] = (f16) in[k*N + n]   (W is KxN row-major -> out is NxK row-major)
__global__ __launch_bounds__(256) void transpose_cast_kernel(const float* __restrict__ in,
                                                             _Float16* __restrict__ out,
                                                             int K, int N) {
  long i = (long)blockIdx.x * 256 + threadIdx.x;
  long total = (long)N * K;
  if (i < total) {
    int n = (int)(i / K);
    int k = (int)(i % K);
    out[(long)n * K + k] = (_Float16)in[(long)k * N + n];
  }
}

// ---------------------------------------------------------------- WMMA GEMM
// C[M,N] = A[M,K] (f16, row-major) * Bt[N,K]^T (f16, row-major N x K), f32 accum.
// Double-buffered LDS; global->LDS via async-to-LDS DMA (ASYNCcnt).
// mode 1: C = softplus(C + bias[col]).  Requires M % 128 == 0, K % 64 == 0.
__global__ __launch_bounds__(256) void wmma_gemm_kernel(
    const _Float16* __restrict__ A,
    const _Float16* __restrict__ Bt,
    float* __restrict__ C,
    int M, int N, int K,
    const float* __restrict__ bias, int mode)
{
  __shared__ _Float16 As[2 * LDS_MAT];
  __shared__ _Float16 Bs[2 * LDS_MAT];

  const int tid  = threadIdx.x;
  const int lane = tid & 31;
  const int wave = tid >> 5;
  const int wm   = wave >> 2;      // 0..1 : 64-row band
  const int wn   = wave & 3;       // 0..3 : 32-col band
  const int blockM = blockIdx.y * 128;
  const int blockN = blockIdx.x * 128;

  const int lrow = lane & 15;      // row (A) / col (B) within tile
  const int lgrp = lane >> 4;      // lane group 0/1
  const int lkgA = lgrp * 8;       // A frag K base (ISA 16-bit A 16x32 layout)

  // per-thread async-copy slot: 8 halves (16B); 32 rows per issue, 4 issues/matrix
  const int ldRow0 = tid >> 3;     // 0..31
  const int ldCol  = (tid & 7) * 8;

  v8f acc[4][2] = {};

  // pre-zero out-of-range B rows in both buffers (those lanes never async-load)
  if (blockN + 128 > N) {
    if (tid < 128 && (blockN + tid) >= N) {
      #pragma unroll
      for (int c2 = 0; c2 < LDS_K; c2 += 8) {
        *(v8h*)(&Bs[0 * LDS_MAT + tid * LDS_STRIDE + c2]) = {};
        *(v8h*)(&Bs[1 * LDS_MAT + tid * LDS_STRIDE + c2]) = {};
      }
    }
  }

  auto issue_stage = [&](int k0, int buf) {
    #pragma unroll
    for (int it = 0; it < 4; ++it) {
      int r = it * 32 + ldRow0;
      unsigned lo = (unsigned)(size_t)&As[buf * LDS_MAT + r * LDS_STRIDE + ldCol];
      const _Float16* gp = A + (long)(blockM + r) * K + k0 + ldCol;
      asm volatile("global_load_async_to_lds_b128 %0, %1, off"
                   :: "v"(lo), "v"(gp) : "memory");
    }
    #pragma unroll
    for (int it = 0; it < 4; ++it) {
      int r  = it * 32 + ldRow0;
      int gn = blockN + r;
      unsigned lo = (unsigned)(size_t)&Bs[buf * LDS_MAT + r * LDS_STRIDE + ldCol];
      const _Float16* gp = Bt + (long)gn * K + k0 + ldCol;
      if (gn < N)
        asm volatile("global_load_async_to_lds_b128 %0, %1, off"
                     :: "v"(lo), "v"(gp) : "memory");
    }
  };

  const int nstages = K / LDS_K;
  issue_stage(0, 0);

  for (int s = 0; s < nstages; ++s) {
    asm volatile("s_wait_asynccnt 0x0" ::: "memory");
    __syncthreads();                       // all tiles of stage s landed; buf (s+1)&1 free
    if (s + 1 < nstages) issue_stage((s + 1) * LDS_K, (s + 1) & 1);

    const _Float16* Ab = &As[(s & 1) * LDS_MAT];
    const _Float16* Bb = &Bs[(s & 1) * LDS_MAT];

    #pragma unroll
    for (int ks = 0; ks < LDS_K; ks += 32) {
      v16h af[4], bf[2];
      #pragma unroll
      for (int i = 0; i < 4; ++i) {
        const _Float16* p = &Ab[(wm * 64 + i * 16 + lrow) * LDS_STRIDE + ks + lkgA];
        v8h lo2 = *(const v8h*)p;
        v8h hi2 = *(const v8h*)(p + 16);
        #pragma unroll
        for (int e = 0; e < 8; ++e) { af[i][e] = lo2[e]; af[i][e + 8] = hi2[e]; }
      }
      #pragma unroll
      for (int j = 0; j < 2; ++j) {
        const _Float16* p = &Bb[(wn * 32 + j * 16 + lrow) * LDS_STRIDE + ks + lgrp * 8];
        v8h lo2 = *(const v8h*)p;
        v8h hi2 = *(const v8h*)(p + 16);
        #pragma unroll
        for (int e = 0; e < 8; ++e) { bf[j][e] = lo2[e]; bf[j][e + 8] = hi2[e]; }
      }
      #pragma unroll
      for (int i = 0; i < 4; ++i)
        #pragma unroll
        for (int j = 0; j < 2; ++j)
          acc[i][j] = __builtin_amdgcn_wmma_f32_16x16x32_f16(
              false, af[i], false, bf[j], (short)0, acc[i][j], false, false);
    }
  }

  // C/D layout: VGPR v -> M = v + lgrp*8 within tile; N = lrow
  #pragma unroll
  for (int i = 0; i < 4; ++i)
    #pragma unroll
    for (int j = 0; j < 2; ++j)
      #pragma unroll
      for (int v = 0; v < 8; ++v) {
        int row = blockM + wm * 64 + i * 16 + lgrp * 8 + v;
        int col = blockN + wn * 32 + j * 16 + lrow;
        if (col < N) {
          float x = acc[i][j][v];
          if (mode == 1) {
            x += bias[col];
            x = (x > 20.f) ? x : log1pf(__expf(x));
          }
          C[(long)row * N + col] = x;
        }
      }
}

// --------------------------------------------- depthwise conv3 + bias + SiLU
__global__ __launch_bounds__(256) void conv_silu_kernel(
    const float* __restrict__ xz, const float* __restrict__ w, const float* __restrict__ bvec,
    float* __restrict__ xc, _Float16* __restrict__ xc_h,
    int L, int d, int dxz, long total)
{
  long idx = (long)blockIdx.x * 256 + threadIdx.x;
  if (idx >= total) return;
  int r = (int)(idx / d);
  int c = (int)(idx % d);
  int t = r % L;
  float x0  = xz[(long)r * dxz + c];
  float xm1 = (t > 0)     ? xz[(long)(r - 1) * dxz + c] : 0.f;
  float xp1 = (t < L - 1) ? xz[(long)(r + 1) * dxz + c] : 0.f;
  float a = bvec[c] + xm1 * w[c * 3 + 0] + x0 * w[c * 3 + 1] + xp1 * w[c * 3 + 2];
  float s = a / (1.f + __expf(-a));          // SiLU
  xc[idx]   = s;
  xc_h[idx] = (_Float16)s;
}

// ---------------------------------------------- slice x_dbl[:, :R] -> f16
__global__ __launch_bounds__(256) void extract_dt_kernel(const float* __restrict__ x_dbl,
                                                         _Float16* __restrict__ out,
                                                         int R, int W, int n) {
  int i = blockIdx.x * 256 + threadIdx.x;
  if (i < n) {
    int r = i / R, j = i % R;
    out[i] = (_Float16)x_dbl[(long)r * W + j];
  }
}

// ------------------------------------------------------- chunked scan phase 1
// Per (b, chunk g, channel c): P[s] = prod_t a_t ; H[s] = chunk output from h=0.
__global__ __launch_bounds__(256) void scan_p1_kernel(
    const float* __restrict__ x_dbl, const float* __restrict__ xc,
    const float* __restrict__ dt, const float* __restrict__ A_log,
    float* __restrict__ Pb, float* __restrict__ Hb, int L, int d)
{
  __shared__ float Bsh[SCAN_LC][16];
  const int tid = threadIdx.x;
  const int b   = blockIdx.y / SCAN_G;
  const int g   = blockIdx.y % SCAN_G;
  const int c   = blockIdx.x * 256 + tid;

  float Ac[16], h[16], P[16];
  #pragma unroll
  for (int s = 0; s < 16; ++s) { Ac[s] = -__expf(A_log[c * 16 + s]); h[s] = 0.f; P[s] = 1.f; }

  const long row0 = (long)b * L + (long)g * SCAN_LC;
  #pragma unroll
  for (int k = 0; k < 4; ++k) {
    int e = tid + k * 256;
    int rr = e >> 4, ss = e & 15;
    Bsh[rr][ss] = x_dbl[(row0 + rr) * 160 + 128 + ss];
  }
  __syncthreads();

  for (int tt = 0; tt < SCAN_LC; ++tt) {
    long r = row0 + tt;
    float u   = xc[r * d + c];
    float dtv = dt[r * d + c];
    float dtu = dtv * u;
    #pragma unroll
    for (int s = 0; s < 16; ++s) {
      float a = __expf(Ac[s] * dtv);
      h[s] = a * h[s] + dtu * Bsh[tt][s];
      P[s] *= a;
    }
  }
  long o = ((long)(b * SCAN_G + g) * d + c) * 16;
  #pragma unroll
  for (int s = 0; s < 16; ++s) { Pb[o + s] = P[s]; Hb[o + s] = h[s]; }
}

// ------------------------------------------------------- chunked scan phase 2
// Serial combine over G chunks per (b,c,s): cin[g] = state entering chunk g.
__global__ __launch_bounds__(256) void scan_p2_kernel(
    const float* __restrict__ Pb, const float* __restrict__ Hb,
    float* __restrict__ Cin, int d, int n)
{
  int i = blockIdx.x * 256 + threadIdx.x;   // (b, c, s) flat
  if (i >= n) return;
  int b  = i / (d * 16);
  int cs = i % (d * 16);
  float h = 0.f;
  #pragma unroll
  for (int g = 0; g < SCAN_G; ++g) {
    long o = (long)(b * SCAN_G + g) * d * 16 + cs;
    Cin[o] = h;
    h = Pb[o] * h + Hb[o];
  }
}

// ------------------------------------------------------- chunked scan phase 3
// Replay chunk from correct incoming state; y_t = h.C_t; fused gating -> f16.
__global__ __launch_bounds__(256) void scan_p3_kernel(
    const float* __restrict__ x_dbl, const float* __restrict__ xc,
    const float* __restrict__ dt, const float* __restrict__ A_log,
    const float* __restrict__ Cin, const float* __restrict__ Dvec,
    const float* __restrict__ xz, _Float16* __restrict__ y_h,
    int L, int d, int dxz)
{
  __shared__ float Bsh[SCAN_LC][16];
  __shared__ float Csh[SCAN_LC][16];
  const int tid = threadIdx.x;
  const int b   = blockIdx.y / SCAN_G;
  const int g   = blockIdx.y % SCAN_G;
  const int c   = blockIdx.x * 256 + tid;

  float Ac[16], h[16];
  long o = ((long)(b * SCAN_G + g) * d + c) * 16;
  #pragma unroll
  for (int s = 0; s < 16; ++s) {
    Ac[s] = -__expf(A_log[c * 16 + s]);
    h[s]  = Cin[o + s];
  }
  const float Dc = Dvec[c];

  const long row0 = (long)b * L + (long)g * SCAN_LC;
  #pragma unroll
  for (int k = 0; k < 4; ++k) {
    int e = tid + k * 256;
    int rr = e >> 4, ss = e & 15;
    Bsh[rr][ss] = x_dbl[(row0 + rr) * 160 + 128 + ss];
    Csh[rr][ss] = x_dbl[(row0 + rr) * 160 + 144 + ss];
  }
  __syncthreads();

  for (int tt = 0; tt < SCAN_LC; ++tt) {
    long r = row0 + tt;
    float u   = xc[r * d + c];
    float dtv = dt[r * d + c];
    float dtu = dtv * u;
    float yv  = 0.f;
    #pragma unroll
    for (int s = 0; s < 16; ++s) {
      h[s] = __expf(Ac[s] * dtv) * h[s] + dtu * Bsh[tt][s];
      yv  += h[s] * Csh[tt][s];
    }
    float z = xz[r * dxz + d + c];
    float val = (yv + u * Dc) * (z / (1.f + __expf(-z)));
    y_h[r * d + c] = (_Float16)val;
  }
}

// ---------------------------------------------------------------- launcher
extern "C" void kernel_launch(void* const* d_in, const int* in_sizes, int n_in,
                              void* d_out, int out_size, void* d_ws, size_t ws_size,
                              hipStream_t stream) {
  (void)in_sizes; (void)n_in; (void)out_size; (void)ws_size;
  const int B = 2, L = 1024, H = 2048, d = 4096, R = 128;
  const int M   = B * L;          // 2048 tokens
  const int dxz = 2 * d;          // 8192
  const int Wdb = R + 32;         // 160

  const float* x      = (const float*)d_in[0];
  const float* W_in   = (const float*)d_in[1];
  const float* conv_w = (const float*)d_in[2];
  const float* conv_b = (const float*)d_in[3];
  const float* W_x    = (const float*)d_in[4];
  const float* W_dt   = (const float*)d_in[5];
  const float* b_dt   = (const float*)d_in[6];
  const float* A_log  = (const float*)d_in[7];
  const float* Dvec   = (const float*)d_in[8];
  const float* W_out  = (const float*)d_in[9];

  char* ws = (char*)d_ws;
  size_t off = 0;
  auto alloc = [&](size_t bytes) -> char* {
    char* p = ws + off;
    off += (bytes + 255) & ~(size_t)255;
    return p;
  };

  _Float16* x_h     = (_Float16*)alloc((size_t)M * H * 2);
  _Float16* WinT    = (_Float16*)alloc((size_t)dxz * H * 2);     // 8192 x 2048
  float*    xz      = (float*)   alloc((size_t)M * dxz * 4);
  float*    xc      = (float*)   alloc((size_t)M * d * 4);
  _Float16* xc_h    = (_Float16*)alloc((size_t)M * d * 2);
  _Float16* WxT     = (_Float16*)alloc((size_t)Wdb * d * 2);     // 160 x 4096
  float*    x_dbl   = (float*)   alloc((size_t)M * Wdb * 4);
  _Float16* dtin_h  = (_Float16*)alloc((size_t)M * R * 2);
  _Float16* WdtT    = (_Float16*)alloc((size_t)d * R * 2);       // 4096 x 128
  float*    dtf     = (float*)   alloc((size_t)M * d * 4);
  _Float16* y_h     = (_Float16*)alloc((size_t)M * d * 2);
  _Float16* WoutT   = (_Float16*)alloc((size_t)H * d * 2);       // 2048 x 4096
  float*    Pb      = (float*)   alloc((size_t)B * SCAN_G * d * 16 * 4);
  float*    Hb      = (float*)   alloc((size_t)B * SCAN_G * d * 16 * 4);
  float*    Cin     = (float*)   alloc((size_t)B * SCAN_G * d * 16 * 4);

  auto eb = [](long n) { return (int)((n + 255) / 256); };

  // precision prep: casts + weight transposes
  cast_f16_kernel<<<eb((long)M * H), 256, 0, stream>>>(x, x_h, M * H);
  transpose_cast_kernel<<<eb((long)dxz * H), 256, 0, stream>>>(W_in,  WinT,  H, dxz);
  transpose_cast_kernel<<<eb((long)Wdb * d), 256, 0, stream>>>(W_x,   WxT,   d, Wdb);
  transpose_cast_kernel<<<eb((long)d * R),   256, 0, stream>>>(W_dt,  WdtT,  R, d);
  transpose_cast_kernel<<<eb((long)H * d),   256, 0, stream>>>(W_out, WoutT, d, H);

  // GEMM1: xz = x @ W_in   (2048 x 8192 x K2048)
  {
    dim3 g((dxz + 127) / 128, (M + 127) / 128);
    wmma_gemm_kernel<<<g, 256, 0, stream>>>(x_h, WinT, xz, M, dxz, H, nullptr, 0);
  }

  // depthwise conv3 + bias + SiLU
  conv_silu_kernel<<<eb((long)M * d), 256, 0, stream>>>(xz, conv_w, conv_b, xc, xc_h,
                                                        L, d, dxz, (long)M * d);

  // GEMM2: x_dbl = xc @ W_x  (2048 x 160 x K4096)
  {
    dim3 g((Wdb + 127) / 128, (M + 127) / 128);
    wmma_gemm_kernel<<<g, 256, 0, stream>>>(xc_h, WxT, x_dbl, M, Wdb, d, nullptr, 0);
  }

  // dt_in slice -> f16
  extract_dt_kernel<<<eb((long)M * R), 256, 0, stream>>>(x_dbl, dtin_h, R, Wdb, M * R);

  // GEMM3: dt = softplus(dt_in @ W_dt + b_dt)  (2048 x 4096 x K128), fused epilogue
  {
    dim3 g((d + 127) / 128, (M + 127) / 128);
    wmma_gemm_kernel<<<g, 256, 0, stream>>>(dtin_h, WdtT, dtf, M, d, R, b_dt, 1);
  }

  // chunked selective scan (3 phases), gating fused into phase 3
  {
    dim3 g(d / 256, B * SCAN_G);
    scan_p1_kernel<<<g, 256, 0, stream>>>(x_dbl, xc, dtf, A_log, Pb, Hb, L, d);
    scan_p2_kernel<<<eb((long)B * d * 16), 256, 0, stream>>>(Pb, Hb, Cin, d, B * d * 16);
    scan_p3_kernel<<<g, 256, 0, stream>>>(x_dbl, xc, dtf, A_log, Cin, Dvec, xz, y_h,
                                          L, d, dxz);
  }

  // GEMM4: out = y @ W_out  (2048 x 2048 x K4096) -> f32 d_out
  {
    dim3 g((H + 127) / 128, (M + 127) / 128);
    wmma_gemm_kernel<<<g, 256, 0, stream>>>(y_h, WoutT, (float*)d_out, M, H, d, nullptr, 0);
  }
}
